// GAT_71322226917732
// MI455X (gfx1250) — compile-verified, matrix operand order checked by reading, not run on hardware
//
#include <hip/hip_runtime.h>
#include <hip/hip_bf16.h>

// ---------------------------------------------------------------------------
// GATv2 (2 conv layers) + MLP head for MI455X (gfx1250, wave32, WMMA).
// GEMMs use v_wmma_f32_16x16x32_f16; edge phase uses float4 gathers (L2-resident
// xl/xr) + atomic segment max (monotonic uint key) / segment sum / scatter-add.
// ---------------------------------------------------------------------------

typedef __attribute__((ext_vector_type(16))) _Float16 v16h;
typedef __attribute__((ext_vector_type(8)))  _Float16 v8h;
typedef __attribute__((ext_vector_type(8)))  float    v8f;

#define NNODES 50000
#define NEDGES 800000
#define ETOT   (NEDGES + NNODES)
#define FDIM   128
#define NHEAD  4

// Order-preserving float <-> uint key for atomic segment-max.
__device__ __forceinline__ unsigned fkey(float f) {
  unsigned u = __float_as_uint(f);
  return (u & 0x80000000u) ? ~u : (u | 0x80000000u);
}
__device__ __forceinline__ float funkey(unsigned k) {
  unsigned u = (k & 0x80000000u) ? (k & 0x7fffffffu) : ~k;
  return __uint_as_float(u);
}

// ------------------------------ cast f32 -> f16 ----------------------------
__global__ void cast_f16_kernel(const float* __restrict__ in,
                                _Float16* __restrict__ out, int n) {
  int i = blockIdx.x * blockDim.x + threadIdx.x;
  if (i < n) out[i] = (_Float16)in[i];
}

// ------------------------- per-conv accumulator init -----------------------
__global__ void init_conv_kernel(float* __restrict__ acc,
                                 unsigned* __restrict__ mkey,
                                 float* __restrict__ ssum, int n) {
  int i = blockIdx.x * blockDim.x + threadIdx.x;
  if (i < n * FDIM) acc[i] = 0.0f;
  if (i < n * NHEAD) { mkey[i] = 0u; ssum[i] = 0.0f; }
}

// ------------------------------ WMMA GEMM ----------------------------------
// C[M,Nt] = act(A[M,K] (f16) x W[K,Nt] (f32, staged->f16 transposed in LDS) + bias)
// Block: 256 thr (8 waves). Block tile 128(M) x 64(N); wave tile 16 x 64.
// K chunked by 64 (2 wmma k-steps); M,Nt,K assumed multiples of 16/64/64.
#define KC 64
#define BT_STRIDE 72   // halves; 144B row stride, 16B aligned

__global__ __launch_bounds__(256) void wmma_gemm_kernel(
    const _Float16* __restrict__ A, const float* __restrict__ W,
    const float* __restrict__ bias,
    float* __restrict__ outF, _Float16* __restrict__ outH,
    int M, int Nt, int K, int act) {
  __shared__ _Float16 bt[64 * BT_STRIDE];

  const int lane = threadIdx.x & 31;
  const int wave = threadIdx.x >> 5;
  const int half = lane >> 4;      // 0/1: K-interleave half per ISA A/B layout
  const int l15  = lane & 15;      // row (A) / col (B,C) within tile
  const int n0   = blockIdx.y * 64;

  int rowBase = blockIdx.x * 128 + wave * 16;
  if (rowBase > M - 16) rowBase = M - 16;   // M is a multiple of 16

  v8f c[4] = {v8f{}, v8f{}, v8f{}, v8f{}};

  for (int kc = 0; kc < K; kc += KC) {
    // Stage W[kc..kc+63][n0..n0+63] transposed (BT[n][k]) as f16 into LDS.
    {
      int kl = threadIdx.x >> 4;          // 0..15
      const int n4 = (threadIdx.x & 15) * 4;
      #pragma unroll
      for (int p = 0; p < 4; ++p, kl += 16) {
        const float4 wv =
            *(const float4*)(W + (size_t)(kc + kl) * Nt + n0 + n4);
        bt[(n4 + 0) * BT_STRIDE + kl] = (_Float16)wv.x;
        bt[(n4 + 1) * BT_STRIDE + kl] = (_Float16)wv.y;
        bt[(n4 + 2) * BT_STRIDE + kl] = (_Float16)wv.z;
        bt[(n4 + 3) * BT_STRIDE + kl] = (_Float16)wv.w;
      }
    }
    __syncthreads();

    const _Float16* arow = A + (size_t)(rowBase + l15) * K + kc;
    #pragma unroll
    for (int ks = 0; ks < 2; ++ks) {
      const int kb = ks * 32 + half * 8;
      // A fragment (16x32 f16): two contiguous 16B chunks per lane.
      v8h alo = *(const v8h*)(arow + kb);
      v8h ahi = *(const v8h*)(arow + kb + 16);
      v16h a = __builtin_shufflevector(alo, ahi, 0, 1, 2, 3, 4, 5, 6, 7,
                                       8, 9, 10, 11, 12, 13, 14, 15);
      // Batch all four B fragments first (one ds_load clause + single wait),
      // then issue the four WMMAs back-to-back.
      v16h b[4];
      #pragma unroll
      for (int nn = 0; nn < 4; ++nn) {
        const _Float16* bp = bt + (nn * 16 + l15) * BT_STRIDE + kb;
        v8h blo = *(const v8h*)(bp);
        v8h bhi = *(const v8h*)(bp + 16);
        b[nn] = __builtin_shufflevector(blo, bhi, 0, 1, 2, 3, 4, 5, 6, 7,
                                        8, 9, 10, 11, 12, 13, 14, 15);
      }
      #pragma unroll
      for (int nn = 0; nn < 4; ++nn) {
        c[nn] = __builtin_amdgcn_wmma_f32_16x16x32_f16(
            false, a, false, b[nn], (short)0, c[nn], false, false);
      }
    }
    __syncthreads();
  }

  // Epilogue: C layout = lanes 0-15 rows r, lanes 16-31 rows r+8.
  #pragma unroll
  for (int nn = 0; nn < 4; ++nn) {
    const int col = n0 + nn * 16 + l15;
    const float bv = bias ? bias[col] : 0.0f;
    #pragma unroll
    for (int r = 0; r < 8; ++r) {
      const int row = rowBase + half * 8 + r;
      float v = c[nn][r] + bv;
      if (act == 1) v = v > 0.0f ? v : (__expf(v) - 1.0f);        // ELU
      else if (act == 2) v = 1.0f / (1.0f + __expf(-v));          // sigmoid
      if (outF) outF[(size_t)row * Nt + col] = v;
      if (outH) outH[(size_t)row * Nt + col] = (_Float16)v;
    }
  }
}

// ----------------------- edge phase: attention logits ----------------------
__global__ void edge_logits_kernel(const int* __restrict__ ei,
                                   const float* __restrict__ xl,
                                   const float* __restrict__ xr,
                                   const float* __restrict__ att,
                                   float* __restrict__ ebuf,
                                   unsigned* __restrict__ mkey) {
  int idx = blockIdx.x * blockDim.x + threadIdx.x;
  if (idx >= ETOT * NHEAD) return;
  const int e = idx >> 2, h = idx & 3;
  int s, d;
  if (e < NEDGES) { s = ei[e]; d = ei[NEDGES + e]; } else { s = d = e - NEDGES; }

  const float4* pl = (const float4*)(xl + (size_t)s * FDIM + h * 32);
  const float4* pr = (const float4*)(xr + (size_t)d * FDIM + h * 32);
  const float4* pa = (const float4*)(att + h * 32);
  float acc = 0.0f;
  #pragma unroll
  for (int i = 0; i < 8; ++i) {
    const float4 a = pl[i], b = pr[i], w = pa[i];
    float z;
    z = a.x + b.x; z = z > 0.0f ? z : 0.2f * z; acc += z * w.x;
    z = a.y + b.y; z = z > 0.0f ? z : 0.2f * z; acc += z * w.y;
    z = a.z + b.z; z = z > 0.0f ? z : 0.2f * z; acc += z * w.z;
    z = a.w + b.w; z = z > 0.0f ? z : 0.2f * z; acc += z * w.w;
  }
  ebuf[idx] = acc;
  atomicMax(mkey + (size_t)d * NHEAD + h, fkey(acc));
}

// ------------------- edge phase: exp + segment sum (denominator) -----------
__global__ void edge_exp_kernel(const int* __restrict__ ei,
                                float* __restrict__ ebuf,
                                const unsigned* __restrict__ mkey,
                                float* __restrict__ ssum) {
  int idx = blockIdx.x * blockDim.x + threadIdx.x;
  if (idx >= ETOT * NHEAD) return;
  const int e = idx >> 2, h = idx & 3;
  int d;
  if (e < NEDGES) d = ei[NEDGES + e]; else d = e - NEDGES;
  const float m = funkey(mkey[(size_t)d * NHEAD + h]);
  const float p = __expf(ebuf[idx] - m);
  ebuf[idx] = p;
  atomicAdd(ssum + (size_t)d * NHEAD + h, p);
}

// ---------------- edge phase: alpha-weighted scatter aggregation -----------
__global__ void edge_scatter_kernel(const int* __restrict__ ei,
                                    const float* __restrict__ xl,
                                    const float* __restrict__ ebuf,
                                    const float* __restrict__ ssum,
                                    float* __restrict__ acc) {
  int idx = blockIdx.x * blockDim.x + threadIdx.x;
  if (idx >= ETOT * NHEAD) return;
  const int e = idx >> 2, h = idx & 3;
  int s, d;
  if (e < NEDGES) { s = ei[e]; d = ei[NEDGES + e]; } else { s = d = e - NEDGES; }
  const float alpha = ebuf[idx] / ssum[(size_t)d * NHEAD + h];
  const float4* pl = (const float4*)(xl + (size_t)s * FDIM + h * 32);
  float* ob = acc + (size_t)d * FDIM + h * 32;
  #pragma unroll
  for (int i = 0; i < 8; ++i) {
    const float4 a = pl[i];
    atomicAdd(ob + i * 4 + 0, a.x * alpha);
    atomicAdd(ob + i * 4 + 1, a.y * alpha);
    atomicAdd(ob + i * 4 + 2, a.z * alpha);
    atomicAdd(ob + i * 4 + 3, a.w * alpha);
  }
}

// ------------------------- node finish: bias + ELU -> f16 ------------------
__global__ void node_finish_kernel(const float* __restrict__ acc,
                                   const float* __restrict__ b,
                                   _Float16* __restrict__ outH, int n) {
  int i = blockIdx.x * blockDim.x + threadIdx.x;
  if (i >= n * FDIM) return;
  float v = acc[i] + b[i & (FDIM - 1)];
  v = v > 0.0f ? v : (__expf(v) - 1.0f);
  outH[i] = (_Float16)v;
}

// ---------------------------------------------------------------------------
extern "C" void kernel_launch(void* const* d_in, const int* in_sizes, int n_in,
                              void* d_out, int out_size, void* d_ws,
                              size_t ws_size, hipStream_t stream) {
  (void)in_sizes; (void)n_in; (void)out_size; (void)ws_size;
  const float* x    = (const float*)d_in[0];
  const int*   ei   = (const int*)d_in[1];
  const float* Wl1  = (const float*)d_in[2];
  const float* Wr1  = (const float*)d_in[3];
  const float* att1 = (const float*)d_in[4];
  const float* b1   = (const float*)d_in[5];
  const float* Wl2  = (const float*)d_in[6];
  const float* Wr2  = (const float*)d_in[7];
  const float* att2 = (const float*)d_in[8];
  const float* b2   = (const float*)d_in[9];
  const float* W1   = (const float*)d_in[10];
  const float* bb1  = (const float*)d_in[11];
  const float* W2   = (const float*)d_in[12];
  const float* bb2  = (const float*)d_in[13];
  const float* W3   = (const float*)d_in[14];
  const float* bb3  = (const float*)d_in[15];
  float* out = (float*)d_out;

  // -------- workspace carving (256B aligned) --------
  char* ws = (char*)d_ws;
  size_t off = 0;
  auto alloc = [&](size_t bytes) -> char* {
    char* p = ws + off;
    off += (bytes + 255) & ~(size_t)255;
    return p;
  };
  const size_t S_F = (size_t)NNODES * FDIM;
  _Float16* xh  = (_Float16*)alloc(S_F * 2);
  _Float16* h1h = (_Float16*)alloc(S_F * 2);
  _Float16* h2h = (_Float16*)alloc(S_F * 2);
  const size_t uoff = off;                   // union region: conv vs MLP
  float*    xl   = (float*)alloc(S_F * 4);
  float*    xr   = (float*)alloc(S_F * 4);
  float*    accb = (float*)alloc(S_F * 4);
  float*    ebuf = (float*)alloc((size_t)ETOT * NHEAD * 4);
  unsigned* mkey = (unsigned*)alloc((size_t)NNODES * NHEAD * 4);
  float*    ssum = (float*)alloc((size_t)NNODES * NHEAD * 4);
  // MLP intermediates overlay the (then-dead) conv buffers.
  _Float16* m1 = (_Float16*)(ws + uoff);                              // N x 640
  _Float16* m2 = (_Float16*)(ws + uoff + (size_t)NNODES * 640 * 2);   // N x 320

  const int TB = 256;
  const int ngrid = (int)((S_F + TB - 1) / TB);
  const int egrid = (ETOT * NHEAD + TB - 1) / TB;
  const dim3 gconv((NNODES + 127) / 128, 128 / 64);

  // ---- conv1 ----
  cast_f16_kernel<<<ngrid, TB, 0, stream>>>(x, xh, (int)S_F);
  wmma_gemm_kernel<<<gconv, TB, 0, stream>>>(xh, Wl1, nullptr, xl, nullptr,
                                             NNODES, 128, 128, 0);
  wmma_gemm_kernel<<<gconv, TB, 0, stream>>>(xh, Wr1, nullptr, xr, nullptr,
                                             NNODES, 128, 128, 0);
  init_conv_kernel<<<ngrid, TB, 0, stream>>>(accb, mkey, ssum, NNODES);
  edge_logits_kernel<<<egrid, TB, 0, stream>>>(ei, xl, xr, att1, ebuf, mkey);
  edge_exp_kernel<<<egrid, TB, 0, stream>>>(ei, ebuf, mkey, ssum);
  edge_scatter_kernel<<<egrid, TB, 0, stream>>>(ei, xl, ebuf, ssum, accb);
  node_finish_kernel<<<ngrid, TB, 0, stream>>>(accb, b1, h1h, NNODES);

  // ---- conv2 ----
  wmma_gemm_kernel<<<gconv, TB, 0, stream>>>(h1h, Wl2, nullptr, xl, nullptr,
                                             NNODES, 128, 128, 0);
  wmma_gemm_kernel<<<gconv, TB, 0, stream>>>(h1h, Wr2, nullptr, xr, nullptr,
                                             NNODES, 128, 128, 0);
  init_conv_kernel<<<ngrid, TB, 0, stream>>>(accb, mkey, ssum, NNODES);
  edge_logits_kernel<<<egrid, TB, 0, stream>>>(ei, xl, xr, att2, ebuf, mkey);
  edge_exp_kernel<<<egrid, TB, 0, stream>>>(ei, ebuf, mkey, ssum);
  edge_scatter_kernel<<<egrid, TB, 0, stream>>>(ei, xl, ebuf, ssum, accb);
  node_finish_kernel<<<ngrid, TB, 0, stream>>>(accb, b2, h2h, NNODES);

  // ---- MLP head ----
  const dim3 gm1((NNODES + 127) / 128, 640 / 64);
  wmma_gemm_kernel<<<gm1, TB, 0, stream>>>(h2h, W1, bb1, nullptr, m1,
                                           NNODES, 640, 128, 1);
  const dim3 gm2((NNODES + 127) / 128, 320 / 64);
  wmma_gemm_kernel<<<gm2, TB, 0, stream>>>(m1, W2, bb2, nullptr, m2,
                                           NNODES, 320, 640, 1);
  const dim3 gm3((NNODES + 127) / 128, 64 / 64);
  wmma_gemm_kernel<<<gm3, TB, 0, stream>>>(m2, W3, bb3, out, nullptr,
                                           NNODES, 64, 320, 2);
}
